// GlobalSparseAttention_3521873183632
// MI455X (gfx1250) — compile-verified, matrix operand order checked by reading, not run on hardware
//
#include <hip/hip_runtime.h>
#include <stdint.h>

// ---------------------------------------------------------------------------
// Problem constants (from reference)
// ---------------------------------------------------------------------------
#define HEADS   8
#define HD      64          // head dim
#define C_IN    512
#define C_OUT   1536        // 3*C
#define HW_IN   112
#define HW_OUT  56
#define NTOK    (HW_OUT*HW_OUT)   // 3136
#define NT64    (NTOK/64)         // 49
#define KTILES  (NTOK/32)         // 98
#define BATCH   2

typedef __attribute__((ext_vector_type(16))) __bf16 v16bf;
typedef __attribute__((ext_vector_type(8)))  float  v8f;
typedef __attribute__((ext_vector_type(4)))  unsigned int v4u;
typedef __attribute__((ext_vector_type(8)))  int v8i;
typedef __attribute__((ext_vector_type(4)))  int v4i;

union AFrag { v16bf v; unsigned u[8]; };
union CFrag { v8f   v; float    f[8]; };
union BFPair { __bf16 h[2]; unsigned u; };

// K-offset inside a 16-bit 16x32 A/B fragment for VGPR v (0..7), half = lane/16.
// ISA 7.12.2: V0..V3 hold K={0,1..6,7}(+8 for upper half), V4..V7 hold K={16..23}(+8).
__device__ __forceinline__ int frag_k(int v, int half) {
    return 2 * (v & 3) + (v >> 2) * 16 + half * 8;
}

#if __has_builtin(__builtin_amdgcn_tensor_load_to_lds) && __has_builtin(__builtin_amdgcn_s_wait_tensorcnt)
#define USE_TDM 1
#else
#define USE_TDM 0
#endif

// ---------------------------------------------------------------------------
// Kernel 1: convert w_qkv (fp32) -> bf16 (hardware cvt)
// ---------------------------------------------------------------------------
__global__ void convert_w_kernel(const float* __restrict__ w,
                                 __bf16* __restrict__ wbf, int n) {
    int i = blockIdx.x * 256 + threadIdx.x;
    if (i < n) wbf[i] = (__bf16)w[i];
}

// ---------------------------------------------------------------------------
// Kernel 2: fused stride-2 downsample + QKV projection (bf16 WMMA GEMM)
//   qkv[b, o, n] = sum_c w[o,c] * x[b, c, 2*(n/56), 2*(n%56)]
// Output scattered into q/k/v [b][h][n][d] bf16.
// Block = 128 threads (4 waves); tile = 64 (o) x 64 (n); K-chunk = 32.
// ---------------------------------------------------------------------------
__global__ __launch_bounds__(128) void qkv_gemm_kernel(
    const float* __restrict__ x, const __bf16* __restrict__ wbf,
    __bf16* __restrict__ qws, __bf16* __restrict__ kws, __bf16* __restrict__ vws)
{
    __shared__ __bf16 Alds[64][32];   // [o_row][c]  (weights)
    __shared__ __bf16 Blds[64][32];   // [n][c]      (x, transposed)

    const int b     = blockIdx.y;
    const int otile = blockIdx.x / NT64;
    const int ntile = blockIdx.x % NT64;
    const int o0 = otile * 64;
    const int n0 = ntile * 64;

    const int tid  = threadIdx.x;
    const int lane = tid & 31;
    const int wave = tid >> 5;
    const int half = lane >> 4;
    const int l16  = lane & 15;

    CFrag acc[4];
#pragma unroll
    for (int i = 0; i < 4; ++i)
#pragma unroll
        for (int j = 0; j < 8; ++j) acc[i].f[j] = 0.0f;

    for (int kc = 0; kc < C_IN; kc += 32) {
        // --- load A tile (64x32 bf16, contiguous pairs) ---
#pragma unroll
        for (int t = 0; t < 8; ++t) {
            int e  = tid + t * 128;          // e in [0,1024): uint pairs
            int r  = e >> 4;
            int cp = e & 15;
            *(unsigned*)&Alds[r][cp * 2] =
                *(const unsigned*)&wbf[(size_t)(o0 + r) * C_IN + kc + cp * 2];
        }
        // --- load B tile: x downsampled, convert fp32->bf16, transpose to [n][c] ---
#pragma unroll
        for (int t = 0; t < 16; ++t) {
            int e  = tid + t * 128;          // e in [0,2048)
            int cc = e >> 6;                 // channel within chunk
            int n  = e & 63;
            int gn = n0 + n;
            int yy = gn / HW_OUT, xx = gn % HW_OUT;
            float val = x[(((size_t)b * C_IN + kc + cc) * HW_IN + 2 * yy) * HW_IN + 2 * xx];
            Blds[n][cc] = (__bf16)val;
        }
        __syncthreads();

        // --- A fragment: rows m = wave*16 + l16, K per ISA pattern ---
        AFrag af;
#pragma unroll
        for (int vv = 0; vv < 8; ++vv)
            af.u[vv] = *(const unsigned*)&Alds[wave * 16 + l16][frag_k(vv, half)];

#pragma unroll
        for (int fb = 0; fb < 4; ++fb) {
            AFrag bfr;
#pragma unroll
            for (int vv = 0; vv < 8; ++vv)
                bfr.u[vv] = *(const unsigned*)&Blds[fb * 16 + l16][frag_k(vv, half)];
            acc[fb].v = __builtin_amdgcn_wmma_f32_16x16x32_bf16(
                false, af.v, false, bfr.v, (short)0, acc[fb].v, false, false);
        }
        __syncthreads();
    }

    // --- scatter results into q/k/v [b][h][n][d] bf16 ---
    // r = rbase + j for j in [0,8): rbase is a multiple of 8, and 8 | 192, 8 | 64,
    // so h / sel / (d base) are invariant across the unrolled j loop.
    {
        const int rbase = o0 + wave * 16 + 8 * half;
        const int h     = rbase / (3 * HD);
        const int rem   = rbase % (3 * HD);
        const int sel   = rem / HD;
        const int d0    = rem % HD;
        __bf16* dst = (sel == 0) ? qws : (sel == 1) ? kws : vws;
        __bf16* base = dst + (size_t)(b * HEADS + h) * NTOK * HD + d0;
#pragma unroll
        for (int fb = 0; fb < 4; ++fb) {
            const int gn = n0 + fb * 16 + l16;
            // pack 8 consecutive bf16 (16B) -> single wide store
            v4u px;
#pragma unroll
            for (int jj = 0; jj < 4; ++jj) {
                BFPair p;
                p.h[0] = (__bf16)acc[fb].f[2 * jj];
                p.h[1] = (__bf16)acc[fb].f[2 * jj + 1];
                px[jj] = p.u;
            }
            *(v4u*)&base[(size_t)gn * HD] = px;
        }
    }
}

// ---------------------------------------------------------------------------
// Kernel 3: flash attention per (b,h).
// Block = 128 threads (4 waves); 64 queries/block, 16 queries/wave.
// K tiles staged via double-buffered Tensor Data Mover DMA when available.
// ---------------------------------------------------------------------------
__global__ __launch_bounds__(128) void flash_attn_kernel(
    const __bf16* __restrict__ qg, const __bf16* __restrict__ kg,
    const __bf16* __restrict__ vg, float* __restrict__ out)
{
    __shared__ __bf16 Qlds[64][64];        // [query][d]
    __shared__ __bf16 Klds[2][32][64];     // double-buffered [key][d]
    __shared__ __bf16 Vlds[64][32];        // [d][key]  (transposed)
    __shared__ __bf16 Plds[4][16][32];     // per-wave P staging

    const int qt = blockIdx.x;       // query tile (0..48)
    const int bh = blockIdx.y;       // b*HEADS + h (0..15)

    const __bf16* qb = qg + (size_t)bh * NTOK * HD;
    const __bf16* kb = kg + (size_t)bh * NTOK * HD;
    const __bf16* vb = vg + (size_t)bh * NTOK * HD;

    const int tid  = threadIdx.x;
    const int lane = tid & 31;
    const int wave = tid >> 5;
    const int half = lane >> 4;
    const int l16  = lane & 15;

    // --- load Q tile (64x64 bf16) ---
#pragma unroll
    for (int t = 0; t < 16; ++t) {
        int e  = tid + t * 128;              // e in [0,2048): uint pairs
        int r  = e >> 5;
        int cp = e & 31;
        *(unsigned*)&Qlds[r][cp * 2] =
            *(const unsigned*)&qb[(size_t)(qt * 64 + r) * HD + cp * 2];
    }
    __syncthreads();

    // --- Q fragments: 2 chunks over d (16x32 each) ---
    AFrag aq[2];
#pragma unroll
    for (int c = 0; c < 2; ++c)
#pragma unroll
        for (int vv = 0; vv < 8; ++vv)
            aq[c].u[vv] =
                *(const unsigned*)&Qlds[wave * 16 + l16][c * 32 + frag_k(vv, half)];

#if USE_TDM
    // --- TDM descriptor pieces that are loop-invariant (D#, ISA ch.8) ---
    const unsigned lds_k0 = (unsigned)(unsigned long long)(uintptr_t)&Klds[0][0][0];
    const unsigned lds_k1 = (unsigned)(unsigned long long)(uintptr_t)&Klds[1][0][0];
    // group1: data_size=2B; tensor_dim0=64, tensor_dim1=NTOK; tile 64x32; stride0=64
    const v8i g1 = { 0x00010000, (64 << 16), (NTOK << 16), (64 << 16),
                     32, 64, 0, 0 };
    const v4i gz4 = { 0, 0, 0, 0 };
    const v8i gz8 = { 0, 0, 0, 0, 0, 0, 0, 0 };

    // prologue: kick off DMA of K tile 0 into buffer 0
    if (wave == 0) {
        unsigned long long ga = (unsigned long long)(uintptr_t)kb;
        v4u g0 = { 1u, lds_k0, (unsigned)(ga & 0xffffffffull),
                   (unsigned)((ga >> 32) & 0x1ffffffull) | (2u << 30) };
        __builtin_amdgcn_tensor_load_to_lds(g0, g1, gz4, gz4, gz8, 0);
    }
#endif

    // --- online softmax state (rows j + 8*half for this lane's half-wave) ---
    float mrow[8], lrow[8];
    CFrag of[4];
#pragma unroll
    for (int j = 0; j < 8; ++j) { mrow[j] = -3.0e38f; lrow[j] = 0.0f; }
#pragma unroll
    for (int dc = 0; dc < 4; ++dc)
#pragma unroll
        for (int j = 0; j < 8; ++j) of[dc].f[j] = 0.0f;

    for (int kt = 0; kt < KTILES; ++kt) {
        const int buf = kt & 1;
#if !USE_TDM
        // --- fallback: vector-load K tile [key][d] (contiguous pairs) ---
#pragma unroll
        for (int t = 0; t < 8; ++t) {
            int e  = tid + t * 128;          // e in [0,1024): uint pairs
            int ky = e >> 5;
            int cp = e & 31;
            *(unsigned*)&Klds[buf][ky][cp * 2] =
                *(const unsigned*)&kb[(size_t)(kt * 32 + ky) * HD + cp * 2];
        }
#endif
        // --- load V tile transposed to [d][key] ---
#pragma unroll
        for (int t = 0; t < 16; ++t) {
            int e  = tid + t * 128;          // e in [0,2048)
            int d  = e >> 5;
            int ky = e & 31;
            Vlds[d][ky] = vb[(size_t)(kt * 32 + ky) * HD + d];
        }
        if (kt + 1 < KTILES)
            __builtin_prefetch(&vb[(size_t)(kt + 1) * 32 * HD + lane * 64], 0, 1);
#if USE_TDM
        // --- pipeline: issue DMA for tile kt+1, then wait for tile kt.
        // TDM ops of one wave complete in order, so tensorcnt<=1 implies tile kt
        // has landed while tile kt+1 can remain in flight under the compute.
        if (wave == 0) {
            if (kt + 1 < KTILES) {
                unsigned long long ga = (unsigned long long)(uintptr_t)
                    (kb + (size_t)(kt + 1) * 32 * HD);
                v4u g0 = { 1u, (buf ? lds_k0 : lds_k1),
                           (unsigned)(ga & 0xffffffffull),
                           (unsigned)((ga >> 32) & 0x1ffffffull) | (2u << 30) };
                __builtin_amdgcn_tensor_load_to_lds(g0, g1, gz4, gz4, gz8, 0);
                __builtin_amdgcn_s_wait_tensorcnt(1);
            } else {
                __builtin_amdgcn_s_wait_tensorcnt(0);
            }
        }
#endif
        __syncthreads();

        // --- S = Q * K^T : two 16x16 fragments over the 32 keys ---
        CFrag s[2];
#pragma unroll
        for (int nb = 0; nb < 2; ++nb) {
#pragma unroll
            for (int j = 0; j < 8; ++j) s[nb].f[j] = 0.0f;
#pragma unroll
            for (int c = 0; c < 2; ++c) {
                AFrag bk;
#pragma unroll
                for (int vv = 0; vv < 8; ++vv)
                    bk.u[vv] = *(const unsigned*)
                        &Klds[buf][nb * 16 + l16][c * 32 + frag_k(vv, half)];
                s[nb].v = __builtin_amdgcn_wmma_f32_16x16x32_bf16(
                    false, aq[c].v, false, bk.v, (short)0, s[nb].v, false, false);
            }
        }

        // --- online softmax: row max over 32 keys (16-lane xor reduction) ---
        float scl[8];
#pragma unroll
        for (int j = 0; j < 8; ++j) {
            float tm = fmaxf(s[0].f[j], s[1].f[j]);
            tm = fmaxf(tm, __shfl_xor(tm, 1));
            tm = fmaxf(tm, __shfl_xor(tm, 2));
            tm = fmaxf(tm, __shfl_xor(tm, 4));
            tm = fmaxf(tm, __shfl_xor(tm, 8));
            float nm = fmaxf(mrow[j], tm);
            scl[j]   = __expf(mrow[j] - nm);
            mrow[j]  = nm;
        }

        // --- P = exp(S - m); stage to LDS (wave-private, wave-order safe) ---
        float rs[8];
#pragma unroll
        for (int j = 0; j < 8; ++j) rs[j] = 0.0f;
#pragma unroll
        for (int nb = 0; nb < 2; ++nb) {
#pragma unroll
            for (int j = 0; j < 8; ++j) {
                float p = __expf(s[nb].f[j] - mrow[j]);
                rs[j] += p;
                Plds[wave][j + 8 * half][nb * 16 + l16] = (__bf16)p;
            }
        }
#pragma unroll
        for (int j = 0; j < 8; ++j) {
            float r = rs[j];
            r += __shfl_xor(r, 1);
            r += __shfl_xor(r, 2);
            r += __shfl_xor(r, 4);
            r += __shfl_xor(r, 8);
            lrow[j] = lrow[j] * scl[j] + r;
        }

        // --- rescale O accumulators ---
#pragma unroll
        for (int dc = 0; dc < 4; ++dc)
#pragma unroll
            for (int j = 0; j < 8; ++j) of[dc].f[j] *= scl[j];

        // --- reload P as A-fragment (16x32 over the 32 keys) ---
        AFrag pf;
#pragma unroll
        for (int vv = 0; vv < 8; ++vv)
            pf.u[vv] = *(const unsigned*)&Plds[wave][l16][frag_k(vv, half)];

        // --- O += P * V : 4 fragments over d ---
#pragma unroll
        for (int dc = 0; dc < 4; ++dc) {
            AFrag bv;
#pragma unroll
            for (int vv = 0; vv < 8; ++vv)
                bv.u[vv] = *(const unsigned*)&Vlds[dc * 16 + l16][frag_k(vv, half)];
            of[dc].v = __builtin_amdgcn_wmma_f32_16x16x32_bf16(
                false, pf.v, false, bv.v, (short)0, of[dc].v, false, false);
        }
        __syncthreads();
    }

    // --- normalize and write out[b][h*64+d][n] ---
    {
        const int b = bh / HEADS;
        const int h = bh % HEADS;
        const int m0 = wave * 16 + 8 * half;
        float* obase = out + ((size_t)b * C_IN + h * HD) * NTOK + (size_t)qt * 64 + m0;
#pragma unroll
        for (int dc = 0; dc < 4; ++dc) {
            const int d = dc * 16 + l16;
#pragma unroll
            for (int j = 0; j < 8; ++j)
                obase[(size_t)d * NTOK + j] = of[dc].f[j] / lrow[j];
        }
    }
}

// ---------------------------------------------------------------------------
// Host launcher
// ---------------------------------------------------------------------------
extern "C" void kernel_launch(void* const* d_in, const int* in_sizes, int n_in,
                              void* d_out, int out_size, void* d_ws, size_t ws_size,
                              hipStream_t stream) {
    const float* x = (const float*)d_in[0];     // (2, 512, 112, 112) fp32
    const float* w = (const float*)d_in[1];     // (1536, 512) fp32
    float* out = (float*)d_out;                 // (2, 512, 56, 56) fp32

    __bf16* wbf = (__bf16*)d_ws;
    size_t w_elems   = (size_t)C_OUT * C_IN;                 // 786432
    size_t qkv_elems = (size_t)BATCH * HEADS * NTOK * HD;    // 3211264
    __bf16* qws = wbf + w_elems;
    __bf16* kws = qws + qkv_elems;
    __bf16* vws = kws + qkv_elems;

    // 1) weight conversion
    convert_w_kernel<<<(int)((w_elems + 255) / 256), 256, 0, stream>>>(
        w, wbf, (int)w_elems);

    // 2) downsample + QKV projection
    dim3 g1((C_OUT / 64) * NT64, BATCH);
    qkv_gemm_kernel<<<g1, 128, 0, stream>>>(x, wbf, qws, kws, vws);

    // 3) flash attention
    dim3 g2(NT64, BATCH * HEADS);
    flash_attn_kernel<<<g2, 128, 0, stream>>>(qws, kws, vws, out);
}
